// ValuesCoordinatesSelfAttention_41480794144924
// MI455X (gfx1250) — matrix operand order repaired and checked by reading, not compiled, not run on hardware
//
#include <hip/hip_runtime.h>
#include <hip/hip_bf16.h>

typedef __attribute__((ext_vector_type(16))) __bf16 v16bf;
typedef __attribute__((ext_vector_type(8)))  __bf16 v8bf;
typedef __attribute__((ext_vector_type(8)))  float  v8f;

#define NSEQ 2048
#define HEADS 4
#define BATCH 4

// ---------------- WMMA helpers ----------------

__device__ __forceinline__ v8f wmma_bf16(v16bf a, v16bf b, v8f c) {
  return __builtin_amdgcn_wmma_f32_16x16x32_bf16(false, a, false, b, (short)0, c, false, false);
}

// A-matrix 16x32 bf16 tile (ISA 7.12.2): lane holds row m=lane%16;
// lanes 0-15: K = {0..7, 16..23}; lanes 16-31: K = {8..15, 24..31}.
__device__ __forceinline__ v16bf load_tile_a(const __bf16* base, int ld) {
  int lane = threadIdx.x & 31;
  const __bf16* p = base + (size_t)(lane & 15) * ld + ((lane < 16) ? 0 : 8);
  v8bf c0 = *(const v8bf*)p;
  v8bf c1 = *(const v8bf*)(p + 16);
  return __builtin_shufflevector(c0, c1, 0,1,2,3,4,5,6,7,8,9,10,11,12,13,14,15);
}

// B-matrix 32x16 bf16 tile: lane holds column n=lane%16 (a row of the
// N-major transposed source); lanes 0-15: K = {0..7, 16..23}; lanes 16-31: K = {8..15, 24..31}.
__device__ __forceinline__ v16bf load_tile_b(const __bf16* base, int ld) {
  int lane = threadIdx.x & 31;
  const __bf16* p = base + (size_t)(lane & 15) * ld + ((lane < 16) ? 0 : 8);
  v8bf c0 = *(const v8bf*)p;
  v8bf c1 = *(const v8bf*)(p + 16);
  return __builtin_shufflevector(c0, c1, 0,1,2,3,4,5,6,7,8,9,10,11,12,13,14,15);
}

// gfx1250 async copy: 16B global -> LDS, tracked by ASYNCcnt.
// LDS operand = low 32 bits of the generic pointer (LDS offset per aperture map).
__device__ __forceinline__ void async_b128(const __bf16* g, __bf16* l) {
  unsigned loff = (unsigned)(unsigned long long)l;
  asm volatile("global_load_async_to_lds_b128 %0, %1, off"
               :: "v"(loff), "v"(g) : "memory");
}

// ---------------- Elementwise / norm kernels ----------------

// Wt[n*K + k] = (bf16) W[k*N + n]   (transpose + convert)
__global__ __launch_bounds__(256) void wcvt_kernel(const float* __restrict__ W,
                                                   __bf16* __restrict__ Wt, int K, int N) {
  int idx = blockIdx.x * 256 + threadIdx.x;
  if (idx >= K * N) return;
  int n = idx / K, k = idx % K;
  Wt[idx] = (__bf16)W[(size_t)k * N + n];
}

// one wave per row LayerNorm, f32 in -> bf16 out
__global__ __launch_bounds__(256) void layernorm_kernel(const float* __restrict__ X,
                                                        const float* __restrict__ g,
                                                        const float* __restrict__ bta,
                                                        __bf16* __restrict__ out, int D) {
  int row  = blockIdx.x * 8 + (threadIdx.x >> 5);
  int lane = threadIdx.x & 31;
  const float* x = X + (size_t)row * D;
  float s = 0.f, ss = 0.f;
  for (int i = lane; i < D; i += 32) { float a = x[i]; s += a; ss += a * a; }
  for (int m = 16; m >= 1; m >>= 1) { s += __shfl_xor(s, m, 32); ss += __shfl_xor(ss, m, 32); }
  float mean = s / (float)D;
  float var  = ss / (float)D - mean * mean;
  float inv  = rsqrtf(var + 1e-5f);
  for (int i = lane; i < D; i += 32)
    out[(size_t)row * D + i] = (__bf16)((x[i] - mean) * inv * g[i] + bta[i]);
}

// ---------------- Generic WMMA GEMM: C(MxN,f32) = A(MxK,bf16) @ Bt(NxK,bf16)^T + bias ----------------

__global__ __launch_bounds__(256) void gemm_f32out_kernel(const __bf16* __restrict__ A,
                                                          const __bf16* __restrict__ Bt,
                                                          const float* __restrict__ bias,
                                                          float* __restrict__ C,
                                                          int M, int N, int K) {
  int w = threadIdx.x >> 5, lane = threadIdx.x & 31;
  int m0 = blockIdx.x * 16;
  int n0 = blockIdx.y * 128 + w * 16;
  if (n0 >= N) return;
  v8f acc = {};
  for (int k = 0; k < K; k += 32)
    acc = wmma_bf16(load_tile_a(A + (size_t)m0 * K + k, K),
                    load_tile_b(Bt + (size_t)n0 * K + k, K), acc);
  int col = n0 + (lane & 15);
  float bv = bias[col];
  int rbase = m0 + ((lane < 16) ? 0 : 8);
#pragma unroll
  for (int j = 0; j < 8; ++j)
    C[(size_t)(rbase + j) * N + col] = acc[j] + bv;
}

// WMMA GEMM writing the concatenated value tensor TRANSPOSED:
// Vt[(b*H + h)*128 + (col%64 + d_off)][nseq] , bf16, contiguous along nseq.
__global__ __launch_bounds__(256) void gemm_vt_kernel(const __bf16* __restrict__ A,
                                                      const __bf16* __restrict__ Bt,
                                                      const float* __restrict__ bias,
                                                      __bf16* __restrict__ Vt,
                                                      int N, int K, int d_off) {
  int w = threadIdx.x >> 5, lane = threadIdx.x & 31;
  int m0 = blockIdx.x * 16;
  int n0 = blockIdx.y * 128 + w * 16;
  if (n0 >= N) return;
  v8f acc = {};
  for (int k = 0; k < K; k += 32)
    acc = wmma_bf16(load_tile_a(A + (size_t)m0 * K + k, K),
                    load_tile_b(Bt + (size_t)n0 * K + k, K), acc);
  int col = n0 + (lane & 15);
  float bv = bias[col];
  int h = col >> 6, d = (col & 63) + d_off;
  int b = m0 >> 11;
  int nn0 = (m0 & (NSEQ - 1)) + ((lane < 16) ? 0 : 8);
  __bf16* dst = Vt + ((size_t)(b * HEADS + h) * 128 + d) * NSEQ + nn0;
  v8bf ov;
#pragma unroll
  for (int j = 0; j < 8; ++j) ov[j] = (__bf16)(acc[j] + bv);
  *(v8bf*)dst = ov;
}

// ---------------- RMS-norm + head-split packing of Q~ / K~ ----------------
// Q~[(b*H+h)*N + n][0:64]  = rms(qv) / sqrt(64), [64:128] = rms(qc)/sqrt(64)
// K~[(b*H+h)*N + n][0:64]  = rms(kv),            [64:128] = rms(kc)
__global__ __launch_bounds__(256) void pack_qk_kernel(const float* __restrict__ qkv,
                                                      const float* __restrict__ qkc,
                                                      const float* __restrict__ rms_v,
                                                      const float* __restrict__ rms_c,
                                                      __bf16* __restrict__ Qt,
                                                      __bf16* __restrict__ Kt) {
  int row  = blockIdx.x * 8 + (threadIdx.x >> 5);
  int lane = threadIdx.x & 31;
  const float* xv = qkv + (size_t)row * 512;
  const float* xc = qkc + (size_t)row * 512;
  float sv = 0.f, sc = 0.f;
  for (int i = lane; i < 512; i += 32) {
    float a = xv[i]; sv += a * a;
    float b = xc[i]; sc += b * b;
  }
  for (int m = 16; m >= 1; m >>= 1) { sv += __shfl_xor(sv, m, 32); sc += __shfl_xor(sc, m, 32); }
  float iv = rsqrtf(sv * (1.f / 512.f) + 1e-6f);
  float ic = rsqrtf(sc * (1.f / 512.f) + 1e-6f);
  int b = row >> 11, nn = row & (NSEQ - 1);
  for (int i = lane; i < 512; i += 32) {
    int h = i >> 7, d = i & 127;
    float qval, kval;
    if (d < 64) {
      int c = h * 64 + d;
      qval = xv[c] * iv * rms_v[c];
      kval = xv[256 + c] * iv * rms_v[256 + c];
    } else {
      int c = h * 64 + (d - 64);
      qval = xc[c] * ic * rms_c[c];
      kval = xc[256 + c] * ic * rms_c[256 + c];
    }
    size_t oidx = ((size_t)(b * HEADS + h) * NSEQ + nn) * 128 + d;
    Qt[oidx] = (__bf16)(qval * 0.125f);  // 1/sqrt(D), D=64
    Kt[oidx] = (__bf16)kval;
  }
}

// ---------------- Flash attention ----------------
// 1 block = 8 waves = 128 query rows of one (b,h); K~/V~ 32-key blocks are
// staged cooperatively into LDS with gfx1250 async loads, double-buffered.

__global__ __launch_bounds__(256) void attn_kernel(const __bf16* __restrict__ Qt,
                                                   const __bf16* __restrict__ Kt,
                                                   const __bf16* __restrict__ Vt,
                                                   __bf16* __restrict__ Ov,
                                                   __bf16* __restrict__ Oc) {
  __shared__ __bf16 Klds[2][32][128];   // 2 x 8KB, keys x dims
  __shared__ __bf16 Vlds[2][128][32];   // 2 x 8KB, dims x keys (from dim-major Vt)
  __shared__ __bf16 plds[8][16 * 32];   // per-wave P staging (C-layout -> A-layout)
  int tid = threadIdx.x;
  int w = tid >> 5, lane = tid & 31;
  int bh = blockIdx.x >> 4;                       // 16 q-blocks of 128 rows per (b,h)
  int m0 = (blockIdx.x & 15) * 128 + w * 16;      // this wave's 16 query rows
  const __bf16* Qb = Qt + ((size_t)bh * NSEQ + m0) * 128;
  const __bf16* Kb = Kt + (size_t)bh * NSEQ * 128;
  const __bf16* Vb = Vt + (size_t)bh * 128 * NSEQ;

  // cooperative async staging of one 32-key block: 4 async b128 per thread
  auto stage = [&](int buf, int kb) {
    const __bf16* kg = Kb + (size_t)kb * 128;     // 8KB contiguous
#pragma unroll
    for (int c = tid; c < 512; c += 256)
      async_b128(kg + c * 8, &Klds[buf][0][0] + c * 8);
#pragma unroll
    for (int c = tid; c < 512; c += 256) {
      int row = c >> 2, off = (c & 3) * 8;
      async_b128(Vb + (size_t)row * NSEQ + kb + off, &Vlds[buf][row][off]);
    }
  };

  v16bf q[4];
#pragma unroll
  for (int i = 0; i < 4; ++i) q[i] = load_tile_a(Qb + i * 32, 128);

  v8f o[8];
#pragma unroll
  for (int g = 0; g < 8; ++g) o[g] = (v8f){};
  float mrow[8], lrow[8];
#pragma unroll
  for (int j = 0; j < 8; ++j) { mrow[j] = -3.0e38f; lrow[j] = 0.f; }

  __bf16* myp = &plds[w][0];
  int col = lane & 15;
  int rofs = (lane < 16) ? 0 : 8;

  stage(0, 0);
  for (int kb = 0; kb < NSEQ; kb += 32) {
    int buf = (kb >> 5) & 1;
    if (kb + 32 < NSEQ) {
      stage(buf ^ 1, kb + 32);                      // safe: prev compute barrier passed
      asm volatile("s_wait_asynccnt 0x4" ::: "memory");  // own current-buf loads done
    } else {
      asm volatile("s_wait_asynccnt 0x0" ::: "memory");
    }
    __syncthreads();                                // current buf published block-wide

    // S(16x32) = Q(16x128) @ K~block^T  (B-tiles from LDS -> ds_load_b128)
    v8f s0 = {}, s1 = {};
#pragma unroll
    for (int ks = 0; ks < 4; ++ks) {
      s0 = wmma_bf16(q[ks], load_tile_b(&Klds[buf][0][ks * 32], 128), s0);
      s1 = wmma_bf16(q[ks], load_tile_b(&Klds[buf][16][ks * 32], 128), s1);
    }
    // online softmax, rows distributed: VGPR j -> row j+rofs, col = lane%16
#pragma unroll
    for (int j = 0; j < 8; ++j) {
      float mx = fmaxf(s0[j], s1[j]);
      for (int t = 8; t >= 1; t >>= 1) mx = fmaxf(mx, __shfl_xor(mx, t, 32));
      float mn = fmaxf(mrow[j], mx);
      float corr = __expf(mrow[j] - mn);
      float p0 = __expf(s0[j] - mn);
      float p1 = __expf(s1[j] - mn);
      float rs = p0 + p1;
      for (int t = 8; t >= 1; t >>= 1) rs += __shfl_xor(rs, t, 32);
      lrow[j] = lrow[j] * corr + rs;
      mrow[j] = mn;
#pragma unroll
      for (int g = 0; g < 8; ++g) o[g][j] *= corr;
      int r = j + rofs;
      myp[r * 32 + col]      = (__bf16)p0;
      myp[r * 32 + 16 + col] = (__bf16)p1;
    }
    asm volatile("s_wait_dscnt 0x0" ::: "memory");
    v16bf a_p = load_tile_a(myp, 32);  // P in A-layout from LDS
    // O(16x128) += P(16x32) @ V~block(32x128)
#pragma unroll
    for (int g = 0; g < 8; ++g)
      o[g] = wmma_bf16(a_p, load_tile_b(&Vlds[buf][g * 16][0], 32), o[g]);

    __syncthreads();                                // release buf for next stage()
  }

  // epilogue: normalize, split 128 dims back into values/coords head outputs
  int b = bh >> 2, h = bh & 3;
  int rbase = m0 + rofs;
#pragma unroll
  for (int j = 0; j < 8; ++j) {
    float inv = 1.f / lrow[j];
    size_t orow = ((size_t)b * NSEQ + (rbase + j)) * 256;
#pragma unroll
    for (int g = 0; g < 8; ++g) {
      float val = o[g][j] * inv;
      int d = g * 16 + col;
      if (d < 64) Ov[orow + h * 64 + d]        = (__bf16)val;
      else        Oc[orow + h * 64 + (d - 64)] = (__bf16)val;
    }
  }
}

// ---------------- Host-side launch ----------------

extern "C" void kernel_launch(void* const* d_in, const int* in_sizes, int n_in,
                              void* d_out, int out_size, void* d_ws, size_t ws_size,
                              hipStream_t stream) {
  (void)in_sizes; (void)n_in; (void)out_size; (void)ws_size;
  const float* V     = (const float*)d_in[0];
  const float* C     = (const float*)d_in[1];
  const float* vn_g  = (const float*)d_in[2];
  const float* vn_b  = (const float*)d_in[3];
  const float* cn_g  = (const float*)d_in[4];
  const float* cn_b  = (const float*)d_in[5];
  const float* W_vqk = (const float*)d_in[6];
  const float* b_vqk = (const float*)d_in[7];
  const float* rms_v = (const float*)d_in[8];
  const float* W_cqk = (const float*)d_in[9];
  const float* b_cqk = (const float*)d_in[10];
  const float* rms_c = (const float*)d_in[11];
  const float* W_vv  = (const float*)d_in[12];
  const float* b_vv  = (const float*)d_in[13];
  const float* W_cv  = (const float*)d_in[14];
  const float* b_cv  = (const float*)d_in[15];
  const float* W_ov  = (const float*)d_in[16];
  const float* b_ov  = (const float*)d_in[17];
  const float* W_oc  = (const float*)d_in[18];
  const float* b_oc  = (const float*)d_in[19];

  const size_t ROWS = (size_t)BATCH * NSEQ;  // 8192
  char* wsp = (char*)d_ws;
  auto take = [&](size_t bytes) { char* p = wsp; wsp += (bytes + 255) & ~(size_t)255; return p; };

  __bf16* Vn    = (__bf16*)take(ROWS * 256 * 2);
  __bf16* Cn    = (__bf16*)take(ROWS * 64 * 2);
  __bf16* WvqkT = (__bf16*)take(512 * 256 * 2);
  __bf16* WcqkT = (__bf16*)take(512 * 64 * 2);
  __bf16* WvvT  = (__bf16*)take(256 * 256 * 2);
  __bf16* WcvT  = (__bf16*)take(256 * 64 * 2);
  __bf16* WovT  = (__bf16*)take(256 * 256 * 2);
  __bf16* WocT  = (__bf16*)take(64 * 256 * 2);
  float*  qkv   = (float*)take(ROWS * 512 * 4);
  float*  qkc   = (float*)take(ROWS * 512 * 4);
  __bf16* Qt    = (__bf16*)take((size_t)BATCH * HEADS * NSEQ * 128 * 2);
  __bf16* Kt    = (__bf16*)take((size_t)BATCH * HEADS * NSEQ * 128 * 2);
  __bf16* Vt    = (__bf16*)take((size_t)BATCH * HEADS * 128 * NSEQ * 2);
  __bf16* Ov    = (__bf16*)take(ROWS * 256 * 2);
  __bf16* Oc    = (__bf16*)take(ROWS * 256 * 2);

  // 1) weights -> bf16, transposed to N-major (so WMMA B-tiles load contiguously)
  wcvt_kernel<<<dim3((256 * 512 + 255) / 256), 256, 0, stream>>>(W_vqk, WvqkT, 256, 512);
  wcvt_kernel<<<dim3((64 * 512 + 255) / 256), 256, 0, stream>>>(W_cqk, WcqkT, 64, 512);
  wcvt_kernel<<<dim3((256 * 256 + 255) / 256), 256, 0, stream>>>(W_vv, WvvT, 256, 256);
  wcvt_kernel<<<dim3((64 * 256 + 255) / 256), 256, 0, stream>>>(W_cv, WcvT, 64, 256);
  wcvt_kernel<<<dim3((256 * 256 + 255) / 256), 256, 0, stream>>>(W_ov, WovT, 256, 256);
  wcvt_kernel<<<dim3((256 * 64 + 255) / 256), 256, 0, stream>>>(W_oc, WocT, 256, 64);

  // 2) LayerNorm inputs -> bf16
  layernorm_kernel<<<dim3(ROWS / 8), 256, 0, stream>>>(V, vn_g, vn_b, Vn, 256);
  layernorm_kernel<<<dim3(ROWS / 8), 256, 0, stream>>>(C, cn_g, cn_b, Cn, 64);

  // 3) QK projections (f32 pre-RMS)
  gemm_f32out_kernel<<<dim3(ROWS / 16, 4), 256, 0, stream>>>(Vn, WvqkT, b_vqk, qkv, (int)ROWS, 512, 256);
  gemm_f32out_kernel<<<dim3(ROWS / 16, 4), 256, 0, stream>>>(Cn, WcqkT, b_cqk, qkc, (int)ROWS, 512, 64);

  // 4) row-RMS + head split + concat streams along contraction dim
  pack_qk_kernel<<<dim3(ROWS / 8), 256, 0, stream>>>(qkv, qkc, rms_v, rms_c, Qt, Kt);

  // 5) value projections -> transposed concatenated V~ (dims 0:64 = vv, 64:128 = vc)
  gemm_vt_kernel<<<dim3(ROWS / 16, 2), 256, 0, stream>>>(Vn, WvvT, b_vv, Vt, 256, 256, 0);
  gemm_vt_kernel<<<dim3(ROWS / 16, 2), 256, 0, stream>>>(Cn, WcvT, b_cv, Vt, 256, 64, 64);

  // 6) flash attention (shared softmax over summed scores, async double-buffered K/V)
  attn_kernel<<<dim3(BATCH * HEADS * (NSEQ / 128)), 256, 0, stream>>>(Qt, Kt, Vt, Ov, Oc);

  // 7) output projections -> f32 d_out (out_v flat, then out_c flat)
  float* out_v = (float*)d_out;
  float* out_c = out_v + ROWS * 256;
  gemm_f32out_kernel<<<dim3(ROWS / 16, 2), 256, 0, stream>>>(Ov, WovT, b_ov, out_v, (int)ROWS, 256, 256);
  gemm_f32out_kernel<<<dim3(ROWS / 16, 1), 256, 0, stream>>>(Oc, WocT, b_oc, out_c, (int)ROWS, 64, 256);
}
